// GRACE_83167746720145
// MI455X (gfx1250) — matrix operand order repaired and checked by reading, not compiled, not run on hardware
//
#include <hip/hip_runtime.h>
#include <hip/hip_bf16.h>
#include <math.h>

// ---------------------------------------------------------------------------
// Types for CDNA5 WMMA (wave32): v_wmma_f32_16x16x32_bf16
// ---------------------------------------------------------------------------
typedef __attribute__((ext_vector_type(16))) __bf16 v16bf;
typedef __attribute__((ext_vector_type(8)))  __bf16 v8bf;
typedef __attribute__((ext_vector_type(8)))  float  v8f;

enum { EPI_NONE = 0, EPI_BIAS = 1, EPI_BIAS_ELU = 2 };

// A tile in LDS: f32, pitch 36 floats = 144 B (16B-aligned rows; 36-dword
// bank stride, gcd(36,64)=4 -> 16 distinct banks across 16 lanes).
#define APITCHF 36
// B tile in LDS: bf16 transposed, pitch 40 bf16 = 80 B (16B-aligned rows;
// 20-dword bank stride, gcd(20,64)=4 -> 16 distinct banks).
#define BPITCH  40

// ---------------------------------------------------------------------------
// A staging: 64 rows x 32 k, f32, straight global->LDS.
// Main loop: async DMA (GLOBAL_LOAD_ASYNC_TO_LDS_B128, ASYNCcnt path) —
// each thread issues 2 lane-chunks of 16B. Tail: guarded sync loads+stores.
// Row overrun is clamped (garbage lands only in unstored output rows).
// ---------------------------------------------------------------------------
template <bool TAIL>
__device__ __forceinline__
void stage_A(const float* __restrict__ A, int M, int K, int rowBlock, int kb,
             float (*sAf)[APITCHF], int tid)
{
    #pragma unroll
    for (int i = 0; i < 2; ++i) {
        int idx4 = tid + i * 256;            // 0..511
        int r    = idx4 >> 3;                // 0..63
        int k4   = (idx4 & 7) << 2;          // 0..28
        int grow = rowBlock + r;
        if (grow > M - 1) grow = M - 1;      // clamp (safe: row unstored)
        const float* gp = A + (size_t)grow * K + kb + k4;
        if (!TAIL) {
            unsigned lds = (unsigned)(size_t)&sAf[r][k4];   // LDS byte offset
            asm volatile("global_load_async_to_lds_b128 %0, %1, off"
                         :: "v"(lds), "v"(gp) : "memory");
        } else {
            float4 av;
            av.x = (kb + k4 + 0 < K) ? gp[0] : 0.0f;
            av.y = (kb + k4 + 1 < K) ? gp[1] : 0.0f;
            av.z = (kb + k4 + 2 < K) ? gp[2] : 0.0f;
            av.w = (kb + k4 + 3 < K) ? gp[3] : 0.0f;
            *(float4*)&sAf[r][k4] = av;
        }
    }
}

// ---------------------------------------------------------------------------
// B staging: 32 k x 128 cols, f32 global -> bf16 transposed LDS [col][k].
// Coalesced float4 loads; col overrun clamped (Nc%4==0; cols unstored).
// ---------------------------------------------------------------------------
template <bool TAIL>
__device__ __forceinline__
void stage_B(const float* __restrict__ W, int K, int Nc, int colBlock, int kb,
             __bf16 (*sB)[BPITCH], int tid)
{
    #pragma unroll
    for (int i = 0; i < 4; ++i) {
        int idx4 = tid + i * 256;            // 0..1023
        int k    = idx4 >> 5;                // 0..31
        int c4   = (idx4 & 31) << 2;         // 0..124
        int gc   = colBlock + c4;
        if (gc + 4 > Nc) gc = Nc - 4;        // clamp
        float4 wv;
        if (!TAIL || (kb + k < K)) {
            wv = *(const float4*)(W + (size_t)(kb + k) * Nc + gc);
        } else {
            wv.x = 0.0f; wv.y = 0.0f; wv.z = 0.0f; wv.w = 0.0f;
        }
        sB[c4 + 0][k] = (__bf16)wv.x;
        sB[c4 + 1][k] = (__bf16)wv.y;
        sB[c4 + 2][k] = (__bf16)wv.z;
        sB[c4 + 3][k] = (__bf16)wv.w;
    }
}

// ---------------------------------------------------------------------------
// One K-step of WMMA. All fragments are built FIRST (ds_load_b128 + cvt),
// then the 4 WMMAs issue back-to-back: no VALU writes overlap a WMMA's
// source registers right after it reads them, avoiding the 4-coexec
// BF16-WMMA->VALU hazard NOPs (ISA 7.12.1). Layouts per ISA 7.12.2.
// ---------------------------------------------------------------------------
__device__ __forceinline__
void gemm_compute(v8f acc[4], const float (*sAf)[APITCHF], const __bf16 (*sB)[BPITCH],
                  int wave, int r16, int half)
{
    const int c16 = (wave << 4) + r16;
    v8bf blo = *(const v8bf*)&sB[c16][half * 8];
    v8bf bhi = *(const v8bf*)&sB[c16][16 + half * 8];
    v16bf bfv = __builtin_shufflevector(blo, bhi,
        0, 1, 2, 3, 4, 5, 6, 7, 8, 9, 10, 11, 12, 13, 14, 15);

    v16bf afv[4];
    #pragma unroll
    for (int t = 0; t < 4; ++t) {
        const int row = t * 16 + r16;
        const float4* p0 = (const float4*)&sAf[row][half * 8];
        const float4* p1 = (const float4*)&sAf[row][16 + half * 8];
        float4 a0 = p0[0], a1 = p0[1];
        float4 a2 = p1[0], a3 = p1[1];
        afv[t][0]  = (__bf16)a0.x; afv[t][1]  = (__bf16)a0.y;
        afv[t][2]  = (__bf16)a0.z; afv[t][3]  = (__bf16)a0.w;
        afv[t][4]  = (__bf16)a1.x; afv[t][5]  = (__bf16)a1.y;
        afv[t][6]  = (__bf16)a1.z; afv[t][7]  = (__bf16)a1.w;
        afv[t][8]  = (__bf16)a2.x; afv[t][9]  = (__bf16)a2.y;
        afv[t][10] = (__bf16)a2.z; afv[t][11] = (__bf16)a2.w;
        afv[t][12] = (__bf16)a3.x; afv[t][13] = (__bf16)a3.y;
        afv[t][14] = (__bf16)a3.z; afv[t][15] = (__bf16)a3.w;
    }
    #pragma unroll
    for (int t = 0; t < 4; ++t) {
        acc[t] = __builtin_amdgcn_wmma_f32_16x16x32_bf16(
            false, afv[t], false, bfv, (short)0, acc[t], false, false);
    }
}

// ---------------------------------------------------------------------------
// Row-major GEMM: C[M,Nc] = act(A[M,K] @ W[K,Nc] + bias)
// Block = 256 threads = 8 waves; block tile = 64 rows x 128 cols.
// Double-buffered LDS pipeline: async A-DMA + sync B-stage of step s+1
// overlap the WMMA compute of step s; one s_wait_asynccnt + one barrier
// per iteration.
// ---------------------------------------------------------------------------
__global__ __launch_bounds__(256)
void wmma_gemm_bf16(const float* __restrict__ A, const float* __restrict__ W,
                    const float* __restrict__ bias, float* __restrict__ C,
                    int M, int K, int Nc, int epi)
{
    __shared__ float  sAf[2][64][APITCHF];   // 2 x 9216 B
    __shared__ __bf16 sB[2][128][BPITCH];    // 2 x 10240 B

    const int tid  = threadIdx.x;
    const int lane = tid & 31;
    const int wave = tid >> 5;
    const int r16  = lane & 15;
    const int half = lane >> 4;
    const int rowBlock = blockIdx.x * 64;
    const int colBlock = blockIdx.y * 128;

    v8f acc[4] = {};
    const int kMain = K & ~31;
    const int nMain = kMain >> 5;
    int p = 0;

    if (nMain > 0) {
        // prologue: stage step 0
        stage_A<false>(A, M, K, rowBlock, 0, sAf[0], tid);
        stage_B<false>(W, K, Nc, colBlock, 0, sB[0], tid);
        asm volatile("s_wait_asynccnt 0" ::: "memory");
        __syncthreads();
        // steady state: stage s into p^1 while computing s-1 from p
        for (int s = 1; s < nMain; ++s) {
            stage_A<false>(A, M, K, rowBlock, s << 5, sAf[p ^ 1], tid);
            stage_B<false>(W, K, Nc, colBlock, s << 5, sB[p ^ 1], tid);
            gemm_compute(acc, sAf[p], sB[p], wave, r16, half);
            asm volatile("s_wait_asynccnt 0" ::: "memory");
            __syncthreads();
            p ^= 1;
        }
        if (kMain < K) {
            // tail staging overlapped with last main compute
            stage_A<true>(A, M, K, rowBlock, kMain, sAf[p ^ 1], tid);
            stage_B<true>(W, K, Nc, colBlock, kMain, sB[p ^ 1], tid);
            gemm_compute(acc, sAf[p], sB[p], wave, r16, half);
            __syncthreads();
            p ^= 1;
            gemm_compute(acc, sAf[p], sB[p], wave, r16, half);
        } else {
            gemm_compute(acc, sAf[p], sB[p], wave, r16, half);
        }
    } else if (K > 0) {
        // K < 32: single guarded step
        stage_A<true>(A, M, K, rowBlock, 0, sAf[0], tid);
        stage_B<true>(W, K, Nc, colBlock, 0, sB[0], tid);
        __syncthreads();
        gemm_compute(acc, sAf[0], sB[0], wave, r16, half);
    }

    // ---- epilogue: C layout: VGPR r -> M = r + 8*(lane>>4), N = lane&15 ----
    const int col = colBlock + (wave << 4) + r16;
    float bv = (epi != EPI_NONE && col < Nc) ? bias[col] : 0.0f;
    #pragma unroll
    for (int t = 0; t < 4; ++t) {
        #pragma unroll
        for (int r = 0; r < 8; ++r) {
            int row = rowBlock + t * 16 + r + half * 8;
            float v = acc[t][r] + bv;
            if (epi == EPI_BIAS_ELU) v = (v > 0.0f) ? v : expm1f(v);
            if (row < M && col < Nc) C[(size_t)row * Nc + col] = v;
        }
    }
}

// ---------------------------------------------------------------------------
// Utility kernels
// ---------------------------------------------------------------------------
__global__ void fill_f32(float* __restrict__ p, float v, long long n)
{
    long long i = (long long)blockIdx.x * blockDim.x + threadIdx.x;
    if (i < n) p[i] = v;
}

// per-column sum / sum-of-squares, chunked over rows, merged via f32 atomics
__global__ void colstats(const float* __restrict__ X, int rows, int D,
                         float* __restrict__ stats /* [2*D]: sum, sumsq */)
{
    int col = blockIdx.x * blockDim.x + threadIdx.x;
    if (col >= D) return;
    int nch = gridDim.y;
    int per = (rows + nch - 1) / nch;
    int r0 = blockIdx.y * per;
    int r1 = min(rows, r0 + per);
    float s = 0.0f, q = 0.0f;
    for (int r = r0; r < r1; ++r) {
        float v = X[(size_t)r * D + col];
        s += v; q += v * v;
    }
    __hip_atomic_fetch_add(&stats[col],     s, __ATOMIC_RELAXED, __HIP_MEMORY_SCOPE_AGENT);
    __hip_atomic_fetch_add(&stats[D + col], q, __ATOMIC_RELAXED, __HIP_MEMORY_SCOPE_AGENT);
}

__global__ void bn_finalize(const float* __restrict__ stats, float rows, int D,
                            float* __restrict__ bnp /* [2*D]: mu, rsig */)
{
    int c = blockIdx.x * blockDim.x + threadIdx.x;
    if (c >= D) return;
    float mu  = stats[c] / rows;
    float var = stats[D + c] / rows - mu * mu;
    bnp[c]     = mu;
    bnp[D + c] = rsqrtf(var + 1e-3f);
}

// in-place BN + activation. mode 0: ELU, mode 1: sigmoid. grid.y = row
__global__ void bn_act(float* __restrict__ X, int D,
                       const float* __restrict__ g, const float* __restrict__ beta,
                       const float* __restrict__ bnp, int mode)
{
    int c = blockIdx.x * blockDim.x + threadIdx.x;
    if (c >= D) return;
    size_t i = (size_t)blockIdx.y * D + c;
    float v = (X[i] - bnp[c]) * bnp[D + c] * g[c] + beta[c];
    v = (mode == 0) ? ((v > 0.0f) ? v : expm1f(v))
                    : 1.0f / (1.0f + expf(-v));
    X[i] = v;
}

__global__ void deg_scatter(const int* __restrict__ dst, float* __restrict__ deg, int E_)
{
    int e = blockIdx.x * blockDim.x + threadIdx.x;
    if (e >= E_) return;
    __hip_atomic_fetch_add(&deg[dst[e]], 1.0f, __ATOMIC_RELAXED, __HIP_MEMORY_SCOPE_AGENT);
}

__global__ void rsqrt_inplace(float* __restrict__ p, int n)
{
    int i = blockIdx.x * blockDim.x + threadIdx.x;
    if (i < n) p[i] = rsqrtf(p[i]);
}

// agg[dst, c] += h[src, c] * dis[src] * dis[dst]   (D = 1<<dshift)
__global__ void msg_scatter(const float* __restrict__ h, const float* __restrict__ dis,
                            const int* __restrict__ src, const int* __restrict__ dst,
                            float* __restrict__ agg, int E_, int dshift)
{
    long long gid = (long long)blockIdx.x * blockDim.x + threadIdx.x;
    if (gid >= ((long long)E_ << dshift)) return;
    int e = (int)(gid >> dshift);
    int c = (int)(gid & ((1 << dshift) - 1));
    int s = src[e], d = dst[e];
    float w = dis[s] * dis[d];
    float v = h[((size_t)s << dshift) + c] * w;
    __hip_atomic_fetch_add(&agg[((size_t)d << dshift) + c], v,
                           __ATOMIC_RELAXED, __HIP_MEMORY_SCOPE_AGENT);
}

// h = prelu(agg + h*dis^2 + b)   in place
__global__ void gcn_combine(float* __restrict__ h, const float* __restrict__ agg,
                            const float* __restrict__ dis, const float* __restrict__ b,
                            const float* __restrict__ pa, int nodes, int dshift)
{
    long long gid = (long long)blockIdx.x * blockDim.x + threadIdx.x;
    if (gid >= ((long long)nodes << dshift)) return;
    int n = (int)(gid >> dshift);
    int c = (int)(gid & ((1 << dshift) - 1));
    float d = dis[n];
    float v = agg[gid] + h[gid] * d * d + b[c];
    float a = pa[0];
    h[gid] = (v >= 0.0f) ? v : a * v;
}

// attention softmax over 2 views + weighted fuse. One wave per node (D=64):
// lane l holds cols l and l+32; dot products reduced with shfl_xor butterfly.
__global__ void attn_fuse(const float* __restrict__ e1, const float* __restrict__ e2,
                          const float* __restrict__ aw, float* __restrict__ fused,
                          int nodes)
{
    const int lane = threadIdx.x & 31;
    const int wv   = threadIdx.x >> 5;
    const int n    = blockIdx.x * (blockDim.x >> 5) + wv;
    if (n >= nodes) return;
    const float* p1 = e1 + (size_t)n * 64;
    const float* p2 = e2 + (size_t)n * 64;
    float a0 = aw[lane], a1 = aw[lane + 32];
    float x0 = p1[lane], x1 = p1[lane + 32];
    float y0 = p2[lane], y1 = p2[lane + 32];
    float w1 = x0 * a0 + x1 * a1;
    float w2 = y0 * a0 + y1 * a1;
    #pragma unroll
    for (int off = 16; off > 0; off >>= 1) {
        w1 += __shfl_xor(w1, off, 32);
        w2 += __shfl_xor(w2, off, 32);
    }
    float m  = fmaxf(w1, w2);
    float b1 = expf(w1 - m), b2 = expf(w2 - m);
    float inv = 1.0f / (b1 + b2); b1 *= inv; b2 *= inv;
    float* o = fused + (size_t)n * 64;
    o[lane]      = b1 * x0 + b2 * y0;
    o[lane + 32] = b1 * x1 + b2 * y1;
}

// ---------------------------------------------------------------------------
// Orchestration
// ---------------------------------------------------------------------------
static inline unsigned cdivu(long long a, long long b) { return (unsigned)((a + b - 1) / b); }

extern "C" void kernel_launch(void* const* d_in, const int* in_sizes, int n_in,
                              void* d_out, int out_size, void* d_ws, size_t ws_size,
                              hipStream_t stream)
{
    constexpr int N  = 50000, DI = 3000, DE = 128, DH = 128, DP = 64, E = 800000;
    (void)in_sizes; (void)n_in; (void)out_size; (void)ws_size;

    const float* x       = (const float*)d_in[0];
    const int*   e1      = (const int*)d_in[1];
    const int*   e2      = (const int*)d_in[2];
    const float* enc_W   = (const float*)d_in[3];
    const float* enc_b   = (const float*)d_in[4];
    const float* enc_g   = (const float*)d_in[5];
    const float* enc_be  = (const float*)d_in[6];
    const float* conv1_W = (const float*)d_in[7];
    const float* conv1_b = (const float*)d_in[8];
    const float* conv2_W = (const float*)d_in[9];
    const float* conv2_b = (const float*)d_in[10];
    const float* prelu_a = (const float*)d_in[11];
    const float* fc1_W   = (const float*)d_in[12];
    const float* fc1_b   = (const float*)d_in[13];
    const float* fc2_W   = (const float*)d_in[14];
    const float* fc2_b   = (const float*)d_in[15];
    const float* attn_w  = (const float*)d_in[16];
    const float* dec_W   = (const float*)d_in[17];
    const float* dec_b   = (const float*)d_in[18];
    const float* dec_g   = (const float*)d_in[19];
    const float* dec_be  = (const float*)d_in[20];

    float* out   = (float*)d_out;     // [h1 | h2 | fused | de_z]
    float* h1o   = out;
    float* h2o   = out + (size_t)N * DP;
    float* fused = out + 2 * (size_t)N * DP;
    float* dez   = out + 3 * (size_t)N * DP;

    float* ws     = (float*)d_ws;
    float* z      = ws;                            // N*128
    float* hbuf   = z      + (size_t)N * DE;       // N*128
    float* agg128 = hbuf   + (size_t)N * DE;       // N*128
    float* emb1   = agg128 + (size_t)N * DE;       // N*64
    float* emb2   = emb1   + (size_t)N * DP;       // N*64
    float* proj   = emb2   + (size_t)N * DP;       // N*64
    float* agg64  = proj   + (size_t)N * DP;       // N*64
    float* dis    = agg64  + (size_t)N * DP;       // N
    float* stats  = dis    + N;                    // 2*3000
    float* bnp    = stats  + 2 * DI;               // 2*3000

    const dim3 B256(256);

    // ---- Encoder: z = ELU(BN(x @ enc_W + enc_b)) ----
    wmma_gemm_bf16<<<dim3(cdivu(N, 64), cdivu(DE, 128)), B256, 0, stream>>>(
        x, enc_W, enc_b, z, N, DI, DE, EPI_BIAS);
    fill_f32<<<cdivu(2 * DE, 256), B256, 0, stream>>>(stats, 0.0f, 2 * DE);
    colstats<<<dim3(cdivu(DE, 256), 128), B256, 0, stream>>>(z, N, DE, stats);
    bn_finalize<<<cdivu(DE, 256), B256, 0, stream>>>(stats, (float)N, DE, bnp);
    bn_act<<<dim3(cdivu(DE, 256), N), B256, 0, stream>>>(z, DE, enc_g, enc_be, bnp, 0);

    // ---- Two GCA views ----
    const int* eidx[2]  = { e1, e2 };
    float*     embs[2]  = { emb1, emb2 };
    float*     houts[2] = { h1o, h2o };
    for (int v = 0; v < 2; ++v) {
        const int* src = eidx[v];
        const int* dst = eidx[v] + E;
        // dis = (deg+1)^-1/2
        fill_f32<<<cdivu(N, 256), B256, 0, stream>>>(dis, 1.0f, N);
        deg_scatter<<<cdivu(E, 256), B256, 0, stream>>>(dst, dis, E);
        rsqrt_inplace<<<cdivu(N, 256), B256, 0, stream>>>(dis, N);
        // conv1 (128 -> 128)
        wmma_gemm_bf16<<<dim3(cdivu(N, 64), 1), B256, 0, stream>>>(
            z, conv1_W, nullptr, hbuf, N, DE, DH, EPI_NONE);
        fill_f32<<<cdivu((long long)N * DH, 256), B256, 0, stream>>>(agg128, 0.0f, (long long)N * DH);
        msg_scatter<<<cdivu((long long)E * DH, 256), B256, 0, stream>>>(hbuf, dis, src, dst, agg128, E, 7);
        gcn_combine<<<cdivu((long long)N * DH, 256), B256, 0, stream>>>(hbuf, agg128, dis, conv1_b, prelu_a, N, 7);
        // conv2 (128 -> 64)
        wmma_gemm_bf16<<<dim3(cdivu(N, 64), 1), B256, 0, stream>>>(
            hbuf, conv2_W, nullptr, embs[v], N, DH, DP, EPI_NONE);
        fill_f32<<<cdivu((long long)N * DP, 256), B256, 0, stream>>>(agg64, 0.0f, (long long)N * DP);
        msg_scatter<<<cdivu((long long)E * DP, 256), B256, 0, stream>>>(embs[v], dis, src, dst, agg64, E, 6);
        gcn_combine<<<cdivu((long long)N * DP, 256), B256, 0, stream>>>(embs[v], agg64, dis, conv2_b, prelu_a, N, 6);
        // projection head -> d_out
        wmma_gemm_bf16<<<dim3(cdivu(N, 64), 1), B256, 0, stream>>>(
            embs[v], fc1_W, fc1_b, proj, N, DP, DP, EPI_BIAS_ELU);
        wmma_gemm_bf16<<<dim3(cdivu(N, 64), 1), B256, 0, stream>>>(
            proj, fc2_W, fc2_b, houts[v], N, DP, DP, EPI_BIAS);
    }

    // ---- attention fuse (wave per node) ----
    attn_fuse<<<cdivu(N, 8), B256, 0, stream>>>(emb1, emb2, attn_w, fused, N);

    // ---- decoder: de_z = sigmoid(BN(fused @ dec_W + dec_b)), in place ----
    wmma_gemm_bf16<<<dim3(cdivu(N, 64), cdivu(DI, 128)), B256, 0, stream>>>(
        fused, dec_W, dec_b, dez, N, DP, DI, EPI_BIAS);
    fill_f32<<<cdivu(2 * DI, 256), B256, 0, stream>>>(stats, 0.0f, 2 * DI);
    colstats<<<dim3(cdivu(DI, 256), 128), B256, 0, stream>>>(dez, N, DI, stats);
    bn_finalize<<<cdivu(DI, 256), B256, 0, stream>>>(stats, (float)N, DI, bnp);
    bn_act<<<dim3(cdivu(DI, 256), N), B256, 0, stream>>>(dez, DI, dec_g, dec_be, bnp, 1);
}
